// CustomFasterRCNN_44032004718818
// MI455X (gfx1250) — compile-verified, matrix operand order checked by reading, not run on hardware
//
#include <hip/hip_runtime.h>
#include <math.h>

typedef __attribute__((ext_vector_type(16))) _Float16 v16h;
typedef __attribute__((ext_vector_type(8)))  float    v8f;

#define IMG        800
#define PRE_NMS    2000
#define POST_NMS   1000
#define NMS_THRESH 0.7f
#define MIN_SIZE   0.001f
#define LOG_CLAMP  4.135166556742356f
#define A_TOTAL    159375           /* 3 * 53125 anchors */
#define KDIM       2304             /* 9 taps * 256 channels (tap-major packing) */
#define NKSTEP     (KDIM / 32)      /* 72 k-steps */
#define TILES_PER_BATCH 1662        /* 1250 + 313 + 79 + 20 (Mtile = 32 pixels) */

__device__ __forceinline__ void tile_to_level(int t, int& n, int& pixOff, int& til) {
    if      (t < 1250) { n = 200; pixOff = 0;     til = t;        }
    else if (t < 1563) { n = 100; pixOff = 40000; til = t - 1250; }
    else if (t < 1642) { n = 50;  pixOff = 50000; til = t - 1563; }
    else               { n = 25;  pixOff = 52500; til = t - 1642; }
}

// ---------------------------------------------------------------------------
// Weight pre-pack (once per launch): f32 OIHW -> f16, K reordered tap-major
//   packW[o][j*256 + c] = convW[o*2304 + c*9 + j]
// ---------------------------------------------------------------------------
__global__ void pack_convw_kernel(const float* __restrict__ convW, _Float16* __restrict__ packW) {
    int i = blockIdx.x * blockDim.x + threadIdx.x;
    if (i >= 256 * KDIM) return;
    int o = i / KDIM;
    int r = i - o * KDIM;
    int j = r >> 8;          // tap 0..8
    int c = r & 255;         // channel
    packW[i] = (_Float16)convW[o * KDIM + c * 9 + j];
}

// Combined 1x1 head weights: 16 cols (3 cls + 12 bbox + 1 zero pad) x 256 K, f16
__global__ void pack_headw_kernel(const float* __restrict__ clsW, const float* __restrict__ bbW,
                                  _Float16* __restrict__ packH) {
    int i = blockIdx.x * blockDim.x + threadIdx.x;
    if (i >= 16 * 256) return;
    int col = i >> 8, k = i & 255;
    float v = 0.0f;
    if (col < 3)       v = clsW[col * 256 + k];
    else if (col < 15) v = bbW[(col - 3) * 256 + k];
    packH[i] = (_Float16)v;
}

// ---------------------------------------------------------------------------
// Fused RPN head: 3x3 conv (implicit GEMM, WMMA f16->f32) + ReLU + 1x1 heads
// Mtile = 32 pixels/block; double-buffered LDS; weight slabs staged via
// async global->LDS copies (ASYNCcnt), overlapping compute of the prior step.
// ---------------------------------------------------------------------------
__global__ __launch_bounds__(256)
void rpn_head_kernel(const float* __restrict__ f0, const float* __restrict__ f1,
                     const float* __restrict__ f2, const float* __restrict__ f3,
                     const _Float16* __restrict__ packW, const float* __restrict__ convB,
                     const _Float16* __restrict__ packH,
                     const float* __restrict__ clsB, const float* __restrict__ bbB,
                     float* __restrict__ scoresW, float* __restrict__ deltasW)
{
    __shared__ __align__(32) _Float16 Atile[2][32 * 32];     // 2 x (32 pixels x 32 K)
    __shared__ __align__(32) _Float16 Wslab[2][256 * 32];    // 2 x (256 outch x 32 K)
    __shared__ __align__(32) _Float16 hidden[32 * 256];      // 32 pixels x 256 hidden

    const int tb = blockIdx.x;
    const int b  = tb / TILES_PER_BATCH;
    const int t  = tb % TILES_PER_BATCH;
    int n, pixOff, til;
    tile_to_level(t, n, pixOff, til);
    const float* feat = (t < 1250) ? f0 : (t < 1563) ? f1 : (t < 1642) ? f2 : f3;
    const int npix = n * n;
    const int pixBase = til * 32;
    const float* featB = feat + (size_t)b * 256 * npix;

    const int tid  = threadIdx.x;
    const int lane = tid & 31;
    const int wave = tid >> 5;
    const int row  = lane & 15;   // M-row within tile == N-col within tile
    const int kh   = lane >> 4;   // K-half selector
    const int mi   = wave & 1;    // M tile 0/1 (16 pixels each)
    const int ni   = wave >> 1;   // N group 0..3 (64 channels each)

    // Stage the 256x32 f16 weight slab for k-step base k0 into LDS buffer `buf`
    // via async global->LDS b128 copies (64B per thread, tracked by ASYNCcnt).
    auto stageW = [&](int k0, int buf) {
        const _Float16* gsrc = packW + (size_t)tid * KDIM + k0;
        unsigned long long ga = (unsigned long long)(uintptr_t)gsrc;
        unsigned ldsDst = (unsigned)(uintptr_t)&Wslab[buf][tid * 32];
        #pragma unroll
        for (int q = 0; q < 4; ++q) {
            asm volatile("global_load_async_to_lds_b128 %0, %1, off"
                         :: "v"(ldsDst + 16u * q), "v"(ga + 16ull * q)
                         : "memory");
        }
    };

    // im2col gather of the 32x32 A tile. Tap-major K packing: each k-step
    // lies entirely inside one 3x3 tap, so the address math is pure adds.
    auto stageA = [&](int k0, int buf) {
        const int j  = k0 >> 8;          // tap index 0..8
        const int c0 = k0 & 255;         // base channel
        const int ky = j / 3, kx = j - ky * 3;
        for (int e = tid; e < 1024; e += 256) {
            const int p  = e >> 5;
            const int kk = e & 31;
            const int pi = pixBase + p;
            float v = 0.0f;
            if (pi < npix) {
                const int y = pi / n, x = pi - y * n;
                const int iy = y + ky - 1, ix = x + kx - 1;
                if (iy >= 0 && iy < n && ix >= 0 && ix < n)
                    v = featB[(size_t)(c0 + kk) * npix + (size_t)iy * n + ix];
            }
            Atile[buf][p * 32 + kk] = (_Float16)v;
        }
    };

    v8f acc[4] = {v8f{}, v8f{}, v8f{}, v8f{}};

    // Prologue: stage slab 0, drain, publish.
    stageW(0, 0);
    stageA(0, 0);
    asm volatile("s_wait_asynccnt 0x0" ::: "memory");
    __syncthreads();

    for (int kt = 0; kt < NKSTEP; ++kt) {
        const int buf = kt & 1;

        // Stage k-step kt+1 into the other buffer; overlaps the WMMAs below.
        if (kt + 1 < NKSTEP) {
            stageW((kt + 1) * 32, buf ^ 1);
            stageA((kt + 1) * 32, buf ^ 1);
        }

        // Preload A-frag + all 4 B-frags (one LDS clause), then issue the
        // four WMMAs back-to-back into the matrix pipe.
        v16h a  = *(const v16h*)&Atile[buf][(mi * 16 + row) * 32 + kh * 16];
        v16h b0 = *(const v16h*)&Wslab[buf][(ni * 64 +  0 + row) * 32 + kh * 16];
        v16h b1 = *(const v16h*)&Wslab[buf][(ni * 64 + 16 + row) * 32 + kh * 16];
        v16h b2 = *(const v16h*)&Wslab[buf][(ni * 64 + 32 + row) * 32 + kh * 16];
        v16h b3 = *(const v16h*)&Wslab[buf][(ni * 64 + 48 + row) * 32 + kh * 16];
        acc[0] = __builtin_amdgcn_wmma_f32_16x16x32_f16(false, a, false, b0, (short)0, acc[0], false, false);
        acc[1] = __builtin_amdgcn_wmma_f32_16x16x32_f16(false, a, false, b1, (short)0, acc[1], false, false);
        acc[2] = __builtin_amdgcn_wmma_f32_16x16x32_f16(false, a, false, b2, (short)0, acc[2], false, false);
        acc[3] = __builtin_amdgcn_wmma_f32_16x16x32_f16(false, a, false, b3, (short)0, acc[3], false, false);

        // Drain next-buffer copies and publish before anyone reads them /
        // before this buffer is overwritten next iteration.
        asm volatile("s_wait_asynccnt 0x0" ::: "memory");
        __syncthreads();
    }

    // Epilogue: conv bias + ReLU -> hidden tile (f16) in LDS.
    // C/D layout: element r of lane L is (M = r + 8*(L>>4), N = nb + (L&15)).
    #pragma unroll
    for (int tt = 0; tt < 4; ++tt) {
        const int cT = ni * 64 + tt * 16 + row;
        const float bs = convB[cT];
        #pragma unroll
        for (int r = 0; r < 8; ++r) {
            const int m = mi * 16 + r + 8 * kh;
            float v = acc[tt][r] + bs;
            v = v > 0.0f ? v : 0.0f;
            hidden[m * 256 + cT] = (_Float16)v;
        }
    }
    __syncthreads();

    // Fused 1x1 heads: two 16(pix) x 16(ch) x 256 GEMMs (waves 0 and 1).
    if (wave < 2) {
        v8f h = {};
        #pragma unroll
        for (int kt = 0; kt < 8; ++kt) {
            const int k0 = kt * 32;
            v16h a2 = *(const v16h*)&hidden[(wave * 16 + row) * 256 + k0 + kh * 16];
            v16h w2 = *(const v16h*)(packH + (size_t)row * 256 + k0 + kh * 16);
            h = __builtin_amdgcn_wmma_f32_16x16x32_f16(false, a2, false, w2,
                                                       (short)0, h, false, false);
        }
        const int col = row;                    // output channel 0..15
        #pragma unroll
        for (int r = 0; r < 8; ++r) {
            const int m  = wave * 16 + r + 8 * kh;
            const int pi = pixBase + m;
            if (pi >= npix) continue;
            const int gp = pixOff + pi;         // global pixel index
            const float v = h[r];
            if (col < 3) {
                scoresW[(size_t)b * A_TOTAL + (size_t)gp * 3 + col] = v + clsB[col];
            } else if (col < 15) {
                const int c   = col - 3;
                const int anc = c >> 2, cc = c & 3;
                deltasW[((size_t)b * A_TOTAL + (size_t)gp * 3 + anc) * 4 + cc] = v + bbB[c];
            }
        }
    }
}

// ---------------------------------------------------------------------------
// Top-k (2000) selection: histogram over order-preserving keys + compaction
// ---------------------------------------------------------------------------
__device__ __forceinline__ unsigned scoreKey(float s) {
    unsigned u = __float_as_uint(s);
    return (u & 0x80000000u) ? ~u : (u | 0x80000000u);   // monotone ascending
}

__global__ void init_kernel(unsigned* hist) {
    int i = blockIdx.x * blockDim.x + threadIdx.x;
    if (i < 2048) hist[i] = 0u;
}

__global__ void hist_kernel(const float* __restrict__ scoresW, unsigned* __restrict__ hist) {
    int i = blockIdx.x * blockDim.x + threadIdx.x;
    if (i >= 2 * A_TOTAL) return;
    int b = i / A_TOTAL;
    unsigned key = scoreKey(scoresW[i]);
    atomicAdd(&hist[b * 1024 + (key >> 22)], 1u);
}

__global__ void thresh_kernel(const unsigned* __restrict__ hist, unsigned* __restrict__ misc) {
    int b = threadIdx.x;
    if (b < 2) {
        unsigned cum = 0, above = 0; int tbin = 0;
        for (int bin = 1023; bin >= 0; --bin) {
            unsigned c = hist[b * 1024 + bin];
            tbin = bin; above = cum;
            if (cum + c >= PRE_NMS) break;
            cum += c;
        }
        misc[b * 8 + 0] = (unsigned)tbin;
        misc[b * 8 + 1] = above;
        misc[b * 8 + 2] = 0u;   // count: strictly-above slots
        misc[b * 8 + 3] = 0u;   // count: equal-bin slots
    }
}

__global__ void compact_kernel(const float* __restrict__ scoresW, unsigned* __restrict__ misc,
                               int* __restrict__ topkIdx, float* __restrict__ topkScore) {
    int i = blockIdx.x * blockDim.x + threadIdx.x;
    if (i >= 2 * A_TOTAL) return;
    int b  = i / A_TOTAL;
    int li = i - b * A_TOTAL;
    float s = scoresW[i];
    unsigned bin  = scoreKey(s) >> 22;
    unsigned tbin = misc[b * 8 + 0];
    unsigned above = misc[b * 8 + 1];
    int slot = -1;
    if (bin > tbin) {
        slot = (int)atomicAdd(&misc[b * 8 + 2], 1u);
    } else if (bin == tbin) {
        unsigned e = atomicAdd(&misc[b * 8 + 3], 1u);
        if (e < (unsigned)PRE_NMS - above) slot = (int)(above + e);
    }
    if (slot >= 0 && slot < PRE_NMS) {
        topkIdx[b * 2048 + slot]   = li;
        topkScore[b * 2048 + slot] = s;
    }
}

__global__ __launch_bounds__(1024)
void sort_kernel(int* __restrict__ topkIdx, float* __restrict__ topkScore) {
    __shared__ float s[2048];
    __shared__ int   id[2048];
    const int b = blockIdx.x, tid = threadIdx.x;
    for (int i = tid; i < 2048; i += 1024) {
        s[i]  = (i < PRE_NMS) ? topkScore[b * 2048 + i] : -__builtin_inff();
        id[i] = (i < PRE_NMS) ? topkIdx[b * 2048 + i]   : -1;
    }
    __syncthreads();
    for (int k = 2; k <= 2048; k <<= 1) {
        for (int j = k >> 1; j > 0; j >>= 1) {
            for (int i = tid; i < 2048; i += 1024) {
                int ixj = i ^ j;
                if (ixj > i) {
                    bool up = ((i & k) == 0);
                    float si = s[i], sj = s[ixj];
                    bool sw = up ? (si < sj) : (si > sj);   // descending sort
                    if (sw) {
                        s[i] = sj; s[ixj] = si;
                        int ti = id[i]; id[i] = id[ixj]; id[ixj] = ti;
                    }
                }
            }
            __syncthreads();
        }
    }
    for (int i = tid; i < 2048; i += 1024) {
        topkScore[b * 2048 + i] = s[i];
        topkIdx[b * 2048 + i]   = id[i];
    }
}

// ---------------------------------------------------------------------------
// Box decode + clip + min-size keep
// ---------------------------------------------------------------------------
__global__ void decode_kernel(const int* __restrict__ topkIdx, const float* __restrict__ deltasW,
                              const float* __restrict__ anchors,
                              float* __restrict__ boxes, int* __restrict__ keep) {
    int i = blockIdx.x * blockDim.x + threadIdx.x;
    if (i >= 2 * PRE_NMS) return;
    int b  = i / PRE_NMS;
    int li = i - b * PRE_NMS;
    int idx = topkIdx[b * 2048 + li];
    float* o = boxes + (size_t)i * 4;
    if (idx < 0 || idx >= A_TOTAL) {
        o[0] = o[1] = o[2] = o[3] = 0.0f; keep[i] = 0; return;
    }
    const float* d = deltasW + ((size_t)b * A_TOTAL + idx) * 4;
    const float* a = anchors + (size_t)idx * 4;
    float w  = a[2] - a[0], h = a[3] - a[1];
    float cx = a[0] + 0.5f * w, cy = a[1] + 0.5f * h;
    float dw = fminf(d[2], LOG_CLAMP), dh = fminf(d[3], LOG_CLAMP);
    float pcx = d[0] * w + cx, pcy = d[1] * h + cy;
    float pw = expf(dw) * w, ph = expf(dh) * h;
    float x0 = fminf(fmaxf(pcx - 0.5f * pw, 0.0f), (float)IMG);
    float y0 = fminf(fmaxf(pcy - 0.5f * ph, 0.0f), (float)IMG);
    float x1 = fminf(fmaxf(pcx + 0.5f * pw, 0.0f), (float)IMG);
    float y1 = fminf(fmaxf(pcy + 0.5f * ph, 0.0f), (float)IMG);
    o[0] = x0; o[1] = y0; o[2] = x1; o[3] = y1;
    keep[i] = ((x1 - x0) >= MIN_SIZE) && ((y1 - y0) >= MIN_SIZE);
}

// ---------------------------------------------------------------------------
// Greedy NMS: one block per batch, boxes resident in LDS
// ---------------------------------------------------------------------------
__global__ __launch_bounds__(1024)
void nms_kernel(const float* __restrict__ boxes, int* __restrict__ keep) {
    __shared__ float bx0[PRE_NMS], by0[PRE_NMS], bx1[PRE_NMS], by1[PRE_NMS], bar[PRE_NMS];
    __shared__ unsigned char kp[PRE_NMS];
    const int b = blockIdx.x, tid = threadIdx.x;
    for (int i = tid; i < PRE_NMS; i += 1024) {
        const float* o = boxes + ((size_t)b * PRE_NMS + i) * 4;
        float x0 = o[0], y0 = o[1], x1 = o[2], y1 = o[3];
        bx0[i] = x0; by0[i] = y0; bx1[i] = x1; by1[i] = y1;
        bar[i] = (x1 - x0) * (y1 - y0);
        kp[i]  = (unsigned char)keep[b * PRE_NMS + i];
    }
    __syncthreads();
    for (int i = 0; i < PRE_NMS; ++i) {
        if (kp[i]) {
            float x0 = bx0[i], y0 = by0[i], x1 = bx1[i], y1 = by1[i], ar = bar[i];
            for (int j = i + 1 + tid; j < PRE_NMS; j += 1024) {
                if (!kp[j]) continue;
                float lx = fmaxf(x0, bx0[j]), ly = fmaxf(y0, by0[j]);
                float rx = fminf(x1, bx1[j]), ry = fminf(y1, by1[j]);
                float iw = fmaxf(rx - lx, 0.0f), ih = fmaxf(ry - ly, 0.0f);
                float inter = iw * ih;
                float un = ar + bar[j] - inter;
                float iou = (un > 0.0f) ? inter / un : 0.0f;
                if (iou > NMS_THRESH) kp[j] = 0;
            }
        }
        __syncthreads();
    }
    for (int i = tid; i < PRE_NMS; i += 1024) keep[b * PRE_NMS + i] = kp[i];
}

// ---------------------------------------------------------------------------
// Stable partition (kept first, score order) + write final outputs
// ---------------------------------------------------------------------------
__global__ __launch_bounds__(256)
void gather_kernel(const float* __restrict__ boxes, const int* __restrict__ keep,
                   float* __restrict__ out) {
    __shared__ int order[PRE_NMS];
    const int b = blockIdx.x, tid = threadIdx.x;
    if (tid == 0) {
        int nk = 0;
        for (int i = 0; i < PRE_NMS; ++i) if (keep[b * PRE_NMS + i])  order[nk++] = i;
        for (int i = 0; i < PRE_NMS && nk < PRE_NMS; ++i)
            if (!keep[b * PRE_NMS + i]) order[nk++] = i;
    }
    __syncthreads();
    for (int o = tid; o < POST_NMS; o += 256) {
        const int i = order[o];
        const int v = keep[b * PRE_NMS + i];
        const float* bx = boxes + ((size_t)b * PRE_NMS + i) * 4;
        float* ob = out + ((size_t)b * POST_NMS + o) * 4;
        ob[0] = v ? bx[0] : 0.0f;
        ob[1] = v ? bx[1] : 0.0f;
        ob[2] = v ? bx[2] : 0.0f;
        ob[3] = v ? bx[3] : 0.0f;
        out[2 * POST_NMS * 4 + b * POST_NMS + o] = v ? 1.0f : 0.0f;
    }
}

// ---------------------------------------------------------------------------
extern "C" void kernel_launch(void* const* d_in, const int* in_sizes, int n_in,
                              void* d_out, int out_size, void* d_ws, size_t ws_size,
                              hipStream_t stream) {
    const float* f0      = (const float*)d_in[0];
    const float* f1      = (const float*)d_in[1];
    const float* f2      = (const float*)d_in[2];
    const float* f3      = (const float*)d_in[3];
    const float* anchors = (const float*)d_in[4];
    const float* convW   = (const float*)d_in[5];
    const float* convB   = (const float*)d_in[6];
    const float* clsW    = (const float*)d_in[7];
    const float* clsB    = (const float*)d_in[8];
    const float* bbW     = (const float*)d_in[9];
    const float* bbB     = (const float*)d_in[10];
    float* out = (float*)d_out;

    // Workspace layout (256B aligned slices)
    char* ws = (char*)d_ws;
    size_t off = 0;
    auto carve = [&](size_t bytes) { char* p = ws + off; off = (off + bytes + 255) & ~(size_t)255; return p; };
    float*     scoresW   = (float*)    carve((size_t)2 * A_TOTAL * 4);
    float*     deltasW   = (float*)    carve((size_t)2 * A_TOTAL * 4 * 4);
    _Float16*  packW     = (_Float16*) carve((size_t)256 * KDIM * 2);
    _Float16*  packH     = (_Float16*) carve((size_t)16 * 256 * 2);
    unsigned*  hist      = (unsigned*) carve((size_t)2 * 1024 * 4);
    unsigned*  misc      = (unsigned*) carve((size_t)2 * 8 * 4);
    int*       topkIdx   = (int*)      carve((size_t)2 * 2048 * 4);
    float*     topkScore = (float*)    carve((size_t)2 * 2048 * 4);
    float*     nboxes    = (float*)    carve((size_t)2 * PRE_NMS * 4 * 4);
    int*       keep      = (int*)      carve((size_t)2 * PRE_NMS * 4);
    (void)ws_size; (void)n_in; (void)in_sizes; (void)out_size;

    // 0) Pre-pack weights to f16 (tap-major K order; combined 1x1 head matrix)
    pack_convw_kernel<<<dim3((256 * KDIM + 255) / 256), dim3(256), 0, stream>>>(convW, packW);
    pack_headw_kernel<<<dim3(16), dim3(256), 0, stream>>>(clsW, bbW, packH);

    // 1) Fused conv + heads (WMMA, double-buffered async-staged weight slabs)
    rpn_head_kernel<<<dim3(2 * TILES_PER_BATCH), dim3(256), 0, stream>>>(
        f0, f1, f2, f3, packW, convB, packH, clsB, bbB, scoresW, deltasW);

    // 2) Top-2000 per batch
    init_kernel<<<dim3(8), dim3(256), 0, stream>>>(hist);
    hist_kernel<<<dim3((2 * A_TOTAL + 255) / 256), dim3(256), 0, stream>>>(scoresW, hist);
    thresh_kernel<<<dim3(1), dim3(32), 0, stream>>>(hist, misc);
    compact_kernel<<<dim3((2 * A_TOTAL + 255) / 256), dim3(256), 0, stream>>>(
        scoresW, misc, topkIdx, topkScore);
    sort_kernel<<<dim3(2), dim3(1024), 0, stream>>>(topkIdx, topkScore);

    // 3) Decode + NMS + gather
    decode_kernel<<<dim3((2 * PRE_NMS + 255) / 256), dim3(256), 0, stream>>>(
        topkIdx, deltasW, anchors, nboxes, keep);
    nms_kernel<<<dim3(2), dim3(1024), 0, stream>>>(nboxes, keep);
    gather_kernel<<<dim3(2), dim3(256), 0, stream>>>(nboxes, keep, out);
}
